// HankelMLP_55155970015423
// MI455X (gfx1250) — matrix operand order repaired and checked by reading, not compiled, not run on hardware
//
#include <hip/hip_runtime.h>

// Problem dims (fixed by the reference)
#define BB_  8
#define SS_  4096
#define FF_  512
#define HH_  2048
#define WW_  16
#define KK1  (WW_ * FF_)        // 8192 : layer-1 K (contiguous window slice)
#define LL_  (SS_ - WW_ + 1)    // 4081 : valid output rows
#define HC_  256                // hidden-chunk width (8 chunks of 256 over H=2048)
#define HSTR 264                // LDS row stride (bf16); 528B = 132 dwords -> conflict-free b128 reads

typedef __bf16 bf16_t;
typedef __attribute__((ext_vector_type(16))) __bf16 v16bf;
typedef __attribute__((ext_vector_type(8)))  __bf16 v8bf;
typedef __attribute__((ext_vector_type(8)))  float  v8f;

union Frag { v16bf f; v8bf h[2]; };

__device__ __forceinline__ unsigned short f32_to_bf16(float x) {
  unsigned int u = __float_as_uint(x);
  u += 0x7fffu + ((u >> 16) & 1u);   // round-to-nearest-even
  return (unsigned short)(u >> 16);
}

// ---------- prep: fp32 -> bf16 elementwise ----------
__global__ void k_cvt_bf16(const float* __restrict__ in, unsigned short* __restrict__ out, int n) {
  int i = blockIdx.x * blockDim.x + threadIdx.x;
  int stride = gridDim.x * blockDim.x;
  for (; i < n; i += stride) out[i] = f32_to_bf16(in[i]);
}

// ---------- prep: fp32 [R][C] -> bf16 [C][R] (transpose+convert); R is a power of two ----------
__global__ void k_tr_cvt_bf16(const float* __restrict__ in, unsigned short* __restrict__ out,
                              int n, int log2R, int C) {
  int i = blockIdx.x * blockDim.x + threadIdx.x;
  int stride = gridDim.x * blockDim.x;
  int Rm1 = (1 << log2R) - 1;
  for (; i < n; i += stride) {
    int r = i & Rm1;          // out fastest dim (coalesced writes)
    int c = i >> log2R;
    out[i] = f32_to_bf16(in[(size_t)r * C + c]);
  }
}

// ---------- fused Hankel-MLP: layer1 (K=8192) -> ReLU -> layer2 (K=2048) ----------
__global__ __launch_bounds__(256)
void hankel_mlp_kernel(const bf16_t* __restrict__ xb,    // [B][S][F] bf16
                       const bf16_t* __restrict__ w1t,   // [H][W*F]  bf16 (transposed)
                       const bf16_t* __restrict__ w2t,   // [F][H]    bf16 (transposed)
                       const float*  __restrict__ b1,    // [H]
                       const float*  __restrict__ b2,    // [F]
                       float*        __restrict__ out)   // [B][S][F]
{
  __shared__ unsigned short hbuf[32 * HSTR];             // 32 x 256 bf16 hidden chunk (padded)

  const int l0   = blockIdx.x * 32;                      // row tile (M = 32)
  const int bb   = blockIdx.y;                           // batch
  const int tid  = threadIdx.x;
  const int wv   = tid >> 5;                             // 0..7
  const int lane = tid & 31;
  const int lm   = lane & 15;                            // lane-in-half
  const int lg   = lane >> 4;                            // half select

  const int akoff = lg * 8;                              // A operand: K {g*8..+7, 16+g*8..+23}
  const int bkoff = lg * 8;                              // B operand: K {g*8..+7, 16+g*8..+23}

  // Per-lane A row pointers for the two 16-row m-tiles (clamped at edge; tail rows zeroed at store)
  int r0 = l0 + lm;        r0 = r0 < (LL_ - 1) ? r0 : (LL_ - 1);
  int r1 = l0 + 16 + lm;   r1 = r1 < (LL_ - 1) ? r1 : (LL_ - 1);
  const bf16_t* xbase = xb + (size_t)bb * (SS_ * FF_);
  const bf16_t* aptr0 = xbase + (size_t)r0 * FF_;
  const bf16_t* aptr1 = xbase + (size_t)r1 * FF_;

  v8f yacc[2][4];
  #pragma unroll
  for (int mi = 0; mi < 2; ++mi)
    #pragma unroll
    for (int ni = 0; ni < 4; ++ni) yacc[mi][ni] = (v8f)(0.0f);

  for (int hc = 0; hc < HH_; hc += HC_) {
    // ---- layer 1: this wave computes h[0..32][hc + wv*32 .. +32] (2m x 2n register tile) ----
    v8f c00 = (v8f)(0.0f), c01 = (v8f)(0.0f), c10 = (v8f)(0.0f), c11 = (v8f)(0.0f);
    const bf16_t* bp0 = w1t + (size_t)(hc + wv * 32 + lm) * KK1;
    const bf16_t* bp1 = bp0 + (size_t)16 * KK1;
    #pragma unroll 2
    for (int k = 0; k < KK1; k += 32) {
      Frag a0, a1, b0, b1f;
      a0.h[0]  = *(const v8bf*)(aptr0 + k + akoff);
      a0.h[1]  = *(const v8bf*)(aptr0 + k + akoff + 16);
      a1.h[0]  = *(const v8bf*)(aptr1 + k + akoff);
      a1.h[1]  = *(const v8bf*)(aptr1 + k + akoff + 16);
      b0.h[0]  = *(const v8bf*)(bp0 + k + bkoff);
      b0.h[1]  = *(const v8bf*)(bp0 + k + bkoff + 16);
      b1f.h[0] = *(const v8bf*)(bp1 + k + bkoff);
      b1f.h[1] = *(const v8bf*)(bp1 + k + bkoff + 16);
      c00 = __builtin_amdgcn_wmma_f32_16x16x32_bf16(false, a0.f, false, b0.f,  (short)0, c00, false, false);
      c01 = __builtin_amdgcn_wmma_f32_16x16x32_bf16(false, a0.f, false, b1f.f, (short)0, c01, false, false);
      c10 = __builtin_amdgcn_wmma_f32_16x16x32_bf16(false, a1.f, false, b0.f,  (short)0, c10, false, false);
      c11 = __builtin_amdgcn_wmma_f32_16x16x32_bf16(false, a1.f, false, b1f.f, (short)0, c11, false, false);
    }
    // bias + ReLU + pack bf16 into LDS
    {
      const float bias0 = b1[hc + wv * 32 + lm];
      const float bias1 = b1[hc + wv * 32 + 16 + lm];
      const int col0 = wv * 32 + lm;
      #pragma unroll
      for (int r = 0; r < 8; ++r) {
        const int m0 = r + lg * 8;                       // C/D layout: half selects row+8
        const int m1 = 16 + m0;
        float v;
        v = c00[r] + bias0; v = v > 0.0f ? v : 0.0f; hbuf[m0 * HSTR + col0]      = f32_to_bf16(v);
        v = c01[r] + bias1; v = v > 0.0f ? v : 0.0f; hbuf[m0 * HSTR + col0 + 16] = f32_to_bf16(v);
        v = c10[r] + bias0; v = v > 0.0f ? v : 0.0f; hbuf[m1 * HSTR + col0]      = f32_to_bf16(v);
        v = c11[r] + bias1; v = v > 0.0f ? v : 0.0f; hbuf[m1 * HSTR + col0 + 16] = f32_to_bf16(v);
      }
    }
    __syncthreads();
    // ---- layer 2 partial: yacc += h[32 x 256] * w2t[wv*64..+64][hc..hc+256] ----
    const bf16_t* hrow0 = (const bf16_t*)&hbuf[lm * HSTR];
    const bf16_t* hrow1 = (const bf16_t*)&hbuf[(16 + lm) * HSTR];
    #pragma unroll
    for (int kk = 0; kk < HC_; kk += 32) {
      Frag a20, a21;
      a20.h[0] = *(const v8bf*)(hrow0 + kk + akoff);
      a20.h[1] = *(const v8bf*)(hrow0 + kk + akoff + 16);
      a21.h[0] = *(const v8bf*)(hrow1 + kk + akoff);
      a21.h[1] = *(const v8bf*)(hrow1 + kk + akoff + 16);
      #pragma unroll
      for (int ni = 0; ni < 4; ++ni) {
        Frag bw;
        const bf16_t* wrow = w2t + (size_t)(wv * 64 + ni * 16 + lm) * HH_ + hc;
        bw.h[0] = *(const v8bf*)(wrow + kk + bkoff);
        bw.h[1] = *(const v8bf*)(wrow + kk + bkoff + 16);
        yacc[0][ni] = __builtin_amdgcn_wmma_f32_16x16x32_bf16(false, a20.f, false, bw.f, (short)0, yacc[0][ni], false, false);
        yacc[1][ni] = __builtin_amdgcn_wmma_f32_16x16x32_bf16(false, a21.f, false, bw.f, (short)0, yacc[1][ni], false, false);
      }
    }
    __syncthreads();                                     // hbuf reused next chunk
  }

  // ---- epilogue: +b2, zero the tail padding rows (l >= L), store fp32 ----
  float* obase = out + (size_t)bb * (SS_ * FF_);
  #pragma unroll
  for (int ni = 0; ni < 4; ++ni) {
    const int f = wv * 64 + ni * 16 + lm;
    const float bias = b2[f];
    #pragma unroll
    for (int mi = 0; mi < 2; ++mi) {
      #pragma unroll
      for (int r = 0; r < 8; ++r) {
        const int l = l0 + mi * 16 + r + lg * 8;
        const float v = (l < LL_) ? (yacc[mi][ni][r] + bias) : 0.0f;
        obase[(size_t)l * FF_ + f] = v;
      }
    }
  }
}

extern "C" void kernel_launch(void* const* d_in, const int* in_sizes, int n_in,
                              void* d_out, int out_size, void* d_ws, size_t ws_size,
                              hipStream_t stream) {
  const float* x  = (const float*)d_in[0];   // [B,S,F]
  const float* w1 = (const float*)d_in[1];   // [W*F, H]
  const float* b1 = (const float*)d_in[2];   // [H]
  const float* w2 = (const float*)d_in[3];   // [H, F]
  const float* b2 = (const float*)d_in[4];   // [F]
  float* out = (float*)d_out;                // [B,S,F]

  const int XBN = BB_ * SS_ * FF_;           // 16,777,216
  const int W1N = KK1 * HH_;                 // 16,777,216
  const int W2N = HH_ * FF_;                 //  1,048,576

  unsigned short* xb  = (unsigned short*)d_ws;
  unsigned short* w1t = xb + XBN;
  unsigned short* w2t = w1t + W1N;           // total ws use ~ 69 MB

  // One-time precision/layout conversion (deterministic, re-run every call).
  k_cvt_bf16<<<8192, 256, 0, stream>>>(x, xb, XBN);
  // w1 [8192][2048] -> w1t [2048][8192] : R = 8192 (2^13), C = 2048
  k_tr_cvt_bf16<<<8192, 256, 0, stream>>>(w1, w1t, W1N, 13, HH_);
  // w2 [2048][512]  -> w2t [512][2048]  : R = 2048 (2^11), C = 512
  k_tr_cvt_bf16<<<2048, 256, 0, stream>>>(w2, w2t, W2N, 11, FF_);

  dim3 grid(SS_ / 32, BB_);                  // 128 l-tiles x 8 batches = 1024 WGs
  hankel_mlp_kernel<<<grid, 256, 0, stream>>>((const bf16_t*)xb, (const bf16_t*)w1t,
                                              (const bf16_t*)w2t, b1, b2, out);
}
